// MoBAInterformerAdapter_10720238370923
// MI455X (gfx1250) — matrix operand order repaired and checked by reading, not compiled, not run on hardware
//
#include <hip/hip_runtime.h>
#include <hip/hip_bf16.h>

// ---------------------------------------------------------------------------
// MoBA Interformer adapter for MI455X (gfx1250, wave32, WMMA).
// All heavy math runs through v_wmma_f32_16x16x32_f16 (f16 operands, fp32
// accum). Softmax / masking / LayerNorm stay in fp32. All LDS fragment
// traffic is 16-byte (ds_*_b128) via fragment-contiguous layouts.
// GEMM: 128x64 block tile, each wave computes 32x64 so B-fragments are
// shared across two row slabs (8 WMMA per 12 ds_load_b128 per K-step).
// ---------------------------------------------------------------------------

typedef __attribute__((ext_vector_type(16))) _Float16 v16h;
typedef __attribute__((ext_vector_type(8)))  _Float16 v8h;
typedef __attribute__((ext_vector_type(8)))  float    v8f;

__device__ __forceinline__ v8f wmma_f16(v16h a, v16h b, v8f c) {
  // 8 args: (neg_a, A, neg_b, B, c_mod, C, reuse_a, reuse_b)
  return __builtin_amdgcn_wmma_f32_16x16x32_f16(false, a, false, b,
                                                (short)0, c, false, false);
}

// 8 consecutive f32 (16B-aligned) -> 8 f16  (lowers to 2x global_load_b128)
__device__ __forceinline__ v8h cvt8(const float* __restrict__ p) {
  float4 f0 = ((const float4*)p)[0];
  float4 f1 = ((const float4*)p)[1];
  v8h r;
  r[0] = (_Float16)f0.x; r[1] = (_Float16)f0.y;
  r[2] = (_Float16)f0.z; r[3] = (_Float16)f0.w;
  r[4] = (_Float16)f1.x; r[5] = (_Float16)f1.y;
  r[6] = (_Float16)f1.z; r[7] = (_Float16)f1.w;
  return r;
}

__device__ __forceinline__ v16h cat16(v8h lo, v8h hi) {
  return __builtin_shufflevector(lo, hi, 0, 1, 2, 3, 4, 5, 6, 7,
                                 8, 9, 10, 11, 12, 13, 14, 15);
}

// 16 consecutive f32 -> v16h
__device__ __forceinline__ v16h cvt16(const float* __restrict__ p) {
  return cat16(cvt8(p), cvt8(p + 8));
}

#define NEGV (-1e9f)

// ---------------------------------------------------------------------------
// 1) Per-node importance scores: scores[g] = dot(x[g,:], score_w) + score_b
// ---------------------------------------------------------------------------
__global__ __launch_bounds__(256)
void score_kernel(const float* __restrict__ x, const float* __restrict__ sw,
                  const float* __restrict__ sb, float* __restrict__ scores,
                  int D, int total_rows) {
  int g    = blockIdx.x * 8 + (threadIdx.x >> 5);
  int lane = threadIdx.x & 31;
  if (g >= total_rows) return;
  const float* xr = x + (size_t)g * D;
  float sum = 0.f;
  for (int i = lane; i < D; i += 32) sum += xr[i] * sw[i];
#pragma unroll
  for (int off = 16; off; off >>= 1) sum += __shfl_xor(sum, off);
  if (lane == 0) scores[g] = sum + sb[0];
}

// ---------------------------------------------------------------------------
// 2) Per-batch: mean of scores, count(scores > mean), clipped to [2, S]
// ---------------------------------------------------------------------------
__global__ __launch_bounds__(256)
void stats_kernel(const float* __restrict__ scores, int S, int* __restrict__ cnt) {
  __shared__ float red[8];
  __shared__ int   ired[8];
  __shared__ float smean;
  int b = blockIdx.x, tid = threadIdx.x;
  const float* sr = scores + (size_t)b * S;
  float sum = 0.f;
  for (int i = tid; i < S; i += 256) sum += sr[i];
#pragma unroll
  for (int off = 16; off; off >>= 1) sum += __shfl_xor(sum, off);
  if ((tid & 31) == 0) red[tid >> 5] = sum;
  __syncthreads();
  if (tid == 0) { float t = 0.f; for (int i = 0; i < 8; ++i) t += red[i]; smean = t / S; }
  __syncthreads();
  float mean = smean;
  int c = 0;
  for (int i = tid; i < S; i += 256) if (sr[i] > mean) c++;
#pragma unroll
  for (int off = 16; off; off >>= 1) c += __shfl_xor(c, off);
  if ((tid & 31) == 0) ired[tid >> 5] = c;
  __syncthreads();
  if (tid == 0) {
    int t = 0; for (int i = 0; i < 8; ++i) t += ired[i];
    t = t < 2 ? 2 : (t > S ? S : t);
    cnt[b] = t;
  }
}

__global__ void finalize_topk_kernel(const int* __restrict__ cnt, int B, int C,
                                     int* __restrict__ keff) {
  int dyn = 0;
  for (int i = 0; i < B; ++i) dyn = (cnt[i] > dyn) ? cnt[i] : dyn;
  int topk = (int)(0.8 * 4.0 + 0.2 * (double)dyn);  // SMOOTH*BASE + (1-SMOOTH)*dyn
  *keff = topk < C ? topk : C;
}

// ---------------------------------------------------------------------------
// 3) WMMA GEMM: Y[M,N] = X[M,K] * W[K,N] + bias   (fp32 io, f16 compute)
//    Block: 128 threads = 4 waves; block tile 128x64; K-step 32.
//    Each wave computes 32x64 as two 16-row slabs sharing the B fragments.
//    sA row-major [m][kk]; W staged transposed [n][kk]: all fragment
//    gathers are 2x ds_load_b128.
// ---------------------------------------------------------------------------
__global__ __launch_bounds__(128)
void gemm_f16wmma(const float* __restrict__ X, const float* __restrict__ W,
                  const float* __restrict__ bias, float* __restrict__ Y,
                  int M, int N, int K) {
  __shared__ __align__(16) _Float16 sA[128][32];  // [row m][kk]
  __shared__ __align__(16) _Float16 sBt[64][32];  // [col n][kk]
  int m0 = blockIdx.x * 128, n0 = blockIdx.y * 64;
  int tid = threadIdx.x;
  int lane = tid & 31, wave = tid >> 5;
  int m16 = lane & 15, hi = lane >> 4;

  // staging coordinates
  int ar = tid >> 1, ah = tid & 1;        // A: row (2 rows/thread), 16-half chunk
  int bc = tid & 63, bh = tid >> 6;       // B: col, which 16-k chunk
  const float* aptr0 = X + (size_t)(m0 + ar) * K + ah * 16;
  const float* aptr1 = X + (size_t)(m0 + 64 + ar) * K + ah * 16;
  const float* wptr  = W + (size_t)(bh * 16) * N + n0 + bc;

  v8f acc0[4] = {};
  v8f acc1[4] = {};
  for (int k0 = 0; k0 < K; k0 += 32) {
    // ---- stage A tile 128x32: 16 contiguous f32 -> 2x ds_store_b128 (x2 rows) ----
    {
      v8h h0 = cvt8(aptr0 + k0);
      v8h h1 = cvt8(aptr0 + k0 + 8);
      v8h* dst = (v8h*)&sA[ar][ah * 16];
      dst[0] = h0;
      dst[1] = h1;
      v8h g0 = cvt8(aptr1 + k0);
      v8h g1 = cvt8(aptr1 + k0 + 8);
      v8h* dst2 = (v8h*)&sA[64 + ar][ah * 16];
      dst2[0] = g0;
      dst2[1] = g1;
    }
    // ---- stage B tile transposed: column reads (coalesced across lanes) ----
    {
      const float* wp = wptr + (size_t)k0 * N;
      v8h w0, w1;
#pragma unroll
      for (int e = 0; e < 8; ++e) {
        w0[e] = (_Float16)wp[(size_t)e * N];
        w1[e] = (_Float16)wp[(size_t)(e + 8) * N];
      }
      v8h* dst = (v8h*)&sBt[bc][bh * 16];
      dst[0] = w0;
      dst[1] = w1;
    }
    if (k0 + 32 < K) {
      __builtin_prefetch(aptr0 + k0 + 32, 0, 1);
      __builtin_prefetch(aptr1 + k0 + 32, 0, 1);
    }
    __syncthreads();

    // ---- fragments: all 2x ds_load_b128; B shared across both row slabs ----
    int arow0 = 16 * wave + m16;
    int arow1 = 64 + 16 * wave + m16;
    v16h a0 = cat16(*(const v8h*)&sA[arow0][hi * 8],
                    *(const v8h*)&sA[arow0][16 + hi * 8]);
    v16h a1 = cat16(*(const v8h*)&sA[arow1][hi * 8],
                    *(const v8h*)&sA[arow1][16 + hi * 8]);
#pragma unroll
    for (int nt = 0; nt < 4; ++nt) {
      const v8h* bp = (const v8h*)&sBt[nt * 16 + m16][hi * 8];
      v16h bf = cat16(bp[0], bp[2]);
      acc0[nt] = wmma_f16(a0, bf, acc0[nt]);
      acc1[nt] = wmma_f16(a1, bf, acc1[nt]);
    }
    __syncthreads();
  }
  // epilogue: C layout row = r + 8*hi, col = lane&15
#pragma unroll
  for (int nt = 0; nt < 4; ++nt) {
    int col = n0 + nt * 16 + m16;
    float bv = bias[col];
#pragma unroll
    for (int r = 0; r < 8; ++r) {
      int row0 = m0 + 16 * wave + r + hi * 8;
      Y[(size_t)row0 * N + col] = acc0[nt][r] + bv;
      Y[(size_t)(row0 + 64) * N + col] = acc1[nt][r] + bv;
    }
  }
}

// ---------------------------------------------------------------------------
// 4) Chunk-mean keys: km[b,c,h,dh] = mean_{j<64} k[b, c*64+j, h, dh]
// ---------------------------------------------------------------------------
__global__ __launch_bounds__(256)
void chunkmean_kernel(const float* __restrict__ k, float* __restrict__ km,
                      int B, int S, int H) {
  const int Dh = 64, CS = 64;
  int C = S / CS;
  int idx = blockIdx.x * 256 + threadIdx.x;
  int total = B * C * H * Dh;
  if (idx >= total) return;
  int dh = idx & 63;
  int rest = idx >> 6;
  int h = rest % H; rest /= H;
  int c = rest % C; int b = rest / C;
  const float* kp = k + ((size_t)(b * S + c * CS)) * (H * Dh) + h * Dh + dh;
  float acc = 0.f;
  for (int j = 0; j < CS; ++j) acc += kp[(size_t)j * H * Dh];
  km[idx] = acc * (1.0f / CS);
}

// ---------------------------------------------------------------------------
// 5) Gate + iterative top-k chunk selection -> 32-bit chunk mask per (b,h,s)
// ---------------------------------------------------------------------------
__global__ __launch_bounds__(128)
void gate_topk_kernel(const float* __restrict__ q, const float* __restrict__ km,
                      const int* __restrict__ keff_p, unsigned* __restrict__ selmask,
                      int B, int S, int H) {
  const int C = 32, Dh = 64;
  int g    = blockIdx.x * 4 + (threadIdx.x >> 5);
  int lane = threadIdx.x & 31;
  int total = B * H * S;
  if (g >= total) return;
  int s = g % S;
  int bh = g / S;
  int h = bh % H, b = bh / H;

  const float* qp  = q  + ((size_t)(b * S + s)) * (H * Dh) + h * Dh;
  const float* kmp = km + (((size_t)b * C + lane) * H + h) * Dh;
  float gv = 0.f;
  for (int d = 0; d < Dh; ++d) gv += qp[d] * kmp[d];

  int  qc     = s >> 6;
  bool self_c = (lane == qc);
  bool past   = ((lane + 1) * 64 <= s + 1);
  if (self_c)     gv = 1e9f;   // current chunk always selected
  else if (!past) gv = NEGV;   // no future chunks in gating

  int keff = *keff_p;
  if (keff > C) keff = C;

  unsigned bits = 0;
  float myv = gv;
  for (int kk = 0; kk < keff; ++kk) {
    float v = myv;
    int   idx = lane;
#pragma unroll
    for (int off = 16; off; off >>= 1) {
      float ov = __shfl_xor(v, off);
      int   oi = __shfl_xor(idx, off);
      if (ov > v || (ov == v && oi < idx)) { v = ov; idx = oi; }
    }
    bits |= (1u << idx);
    if (lane == idx) myv = -3.0e38f;
  }
  if (lane == 0) selmask[(size_t)g] = bits;
}

// ---------------------------------------------------------------------------
// 6) MoBA flash attention: one wave per (b, h, 16-query tile).
//    32-key blocks; chunk-mask skipping; fp32 online softmax; WMMA QK^T + PV.
//    V staged transposed [d][kk] so PV B-fragments are 2x ds_load_b128.
// ---------------------------------------------------------------------------
__global__ __launch_bounds__(128)
void moba_flash_kernel(const float* __restrict__ q, const float* __restrict__ k,
                       const float* __restrict__ v, const unsigned* __restrict__ selmask,
                       float* __restrict__ o, int B, int S, int H) {
  const int Dh = 64;
  __shared__ __align__(16) _Float16 sP[4][16][32];   // [wave][row][key]
  __shared__ __align__(16) _Float16 sVt[4][64][32];  // [wave][dim][key]

  int wave = threadIdx.x >> 5, lane = threadIdx.x & 31;
  int m16 = lane & 15, hi = lane >> 4;
  int nQT = S / 16;
  int g = blockIdx.x * 4 + wave;
  if (g >= B * H * nQT) return;
  int qt = g % nQT;
  int bh = g / nQT;
  int h = bh % H, b = bh / H;
  int s0 = qt * 16;

  // Q fragments for Dh slices [0,32) and [32,64): vector loads + cvt
  v16h aq0, aq1;
  {
    const float* qrow = q + ((size_t)(b * S + s0 + m16)) * (H * Dh) + h * Dh;
    aq0 = cat16(cvt8(qrow + hi * 8), cvt8(qrow + 16 + hi * 8));
    aq1 = cat16(cvt8(qrow + 32 + hi * 8), cvt8(qrow + 48 + hi * 8));
  }

  // per-row chunk masks for the 8 rows this lane's C-fragments touch
  unsigned rmask[8];
  unsigned unionm = 0;
#pragma unroll
  for (int r = 0; r < 8; ++r) {
    rmask[r] = selmask[((size_t)(b * H + h)) * S + s0 + r + hi * 8];
    unionm |= rmask[r];
  }
  unionm |= __shfl_xor(unionm, 16);

  float mrow[8], lrow[8];
  v8f oacc[4] = {};
#pragma unroll
  for (int r = 0; r < 8; ++r) { mrow[r] = -3.0e38f; lrow[r] = 0.f; }

  const float scale = 0.125f;  // 1/sqrt(Dh)

  for (int kb = 0; kb < s0 + 16; kb += 32) {
    int c = kb >> 6;
    if (!((unionm >> c) & 1)) continue;

    // ---- scores: two 16x16 tiles; K-fragments via global_load_b128 + cvt ----
    v8f sc0 = {}, sc1 = {};
#pragma unroll
    for (int dblk = 0; dblk < 2; ++dblk) {
      int d0 = dblk * 32;
      const float* kr0 = k + ((size_t)(b * S + kb + m16))      * (H * Dh) + h * Dh + d0 + hi * 8;
      const float* kr1 = k + ((size_t)(b * S + kb + 16 + m16)) * (H * Dh) + h * Dh + d0 + hi * 8;
      v16h b0 = cat16(cvt8(kr0), cvt8(kr0 + 16));
      v16h b1 = cat16(cvt8(kr1), cvt8(kr1 + 16));
      v16h aa = dblk ? aq1 : aq0;
      sc0 = wmma_f16(aa, b0, sc0);
      sc1 = wmma_f16(aa, b1, sc1);
    }

    // ---- mask + fp32 online softmax (row r at VGPR r, row = r+8*hi) ----
#pragma unroll
    for (int r = 0; r < 8; ++r) {
      int srow = s0 + r + hi * 8;
      bool sel = (rmask[r] >> c) & 1;
      int t0 = kb + m16, t1 = kb + 16 + m16;
      float x0 = (sel && t0 <= srow) ? sc0[r] * scale : NEGV;
      float x1 = (sel && t1 <= srow) ? sc1[r] * scale : NEGV;
      float mx = fmaxf(x0, x1);
#pragma unroll
      for (int off = 8; off; off >>= 1) mx = fmaxf(mx, __shfl_xor(mx, off));
      float mnew = fmaxf(mrow[r], mx);
      float corr = __expf(mrow[r] - mnew);
      float e0 = __expf(x0 - mnew);
      float e1 = __expf(x1 - mnew);
      float rs = e0 + e1;
#pragma unroll
      for (int off = 8; off; off >>= 1) rs += __shfl_xor(rs, off);
      lrow[r] = lrow[r] * corr + rs;
      mrow[r] = mnew;
#pragma unroll
      for (int nt = 0; nt < 4; ++nt) oacc[nt][r] *= corr;
      sP[wave][r + hi * 8][m16]      = (_Float16)e0;
      sP[wave][r + hi * 8][16 + m16] = (_Float16)e1;
    }

    // ---- stage V transposed: lane owns dims (lane, lane+32); reads are
    //      coalesced across lanes; writes are 4x ds_store_b128 per dim ----
#pragma unroll
    for (int dd = 0; dd < 2; ++dd) {
      int d = lane + dd * 32;
      const float* vcol = v + ((size_t)(b * S + kb)) * (H * Dh) + h * Dh + d;
      v8h c0, c1, c2, c3;
#pragma unroll
      for (int e = 0; e < 8; ++e) {
        c0[e] = (_Float16)vcol[(size_t)(e)      * (H * Dh)];
        c1[e] = (_Float16)vcol[(size_t)(e + 8)  * (H * Dh)];
        c2[e] = (_Float16)vcol[(size_t)(e + 16) * (H * Dh)];
        c3[e] = (_Float16)vcol[(size_t)(e + 24) * (H * Dh)];
      }
      v8h* dst = (v8h*)&sVt[wave][d][0];
      dst[0] = c0; dst[1] = c1; dst[2] = c2; dst[3] = c3;
    }
    // Wave-local LDS round-trip: make ds stores land before ds loads.
    asm volatile("s_wait_dscnt 0x0" ::: "memory");
    __builtin_amdgcn_wave_barrier();

    // ---- P*V: A = P (16x32) 2x b128; B = V tiles 2x b128 each ----
    v16h ap;
    {
      v8h plo = *(const v8h*)&sP[wave][m16][hi * 8];
      v8h phi = *(const v8h*)&sP[wave][m16][16 + hi * 8];
      ap = cat16(plo, phi);
    }
#pragma unroll
    for (int nt = 0; nt < 4; ++nt) {
      const v8h* vp = (const v8h*)&sVt[wave][nt * 16 + m16][hi * 8];
      v16h bv = cat16(vp[0], vp[2]);
      oacc[nt] = wmma_f16(ap, bv, oacc[nt]);
    }
    __builtin_amdgcn_wave_barrier();
  }

  // ---- epilogue: normalize and store [b, s, h*64 + d] ----
#pragma unroll
  for (int r = 0; r < 8; ++r) {
    float inv = 1.0f / lrow[r];
    int srow = s0 + r + hi * 8;
    float* orow = o + ((size_t)(b * S + srow)) * (H * Dh) + h * Dh;
#pragma unroll
    for (int nt = 0; nt < 4; ++nt) orow[nt * 16 + m16] = oacc[nt][r] * inv;
  }
}

// ---------------------------------------------------------------------------
// 7) Row LayerNorm (optionally fused ReLU). In-place safe.
// ---------------------------------------------------------------------------
__global__ __launch_bounds__(256)
void layernorm_kernel(const float* __restrict__ x, const float* __restrict__ gam,
                      const float* __restrict__ bet, float* __restrict__ y,
                      int D, int doRelu) {
  __shared__ float red[8];
  __shared__ float smean, svar;
  int row = blockIdx.x, tid = threadIdx.x;
  const float* xr = x + (size_t)row * D;
  float* yr = y + (size_t)row * D;

  float sum = 0.f;
  for (int i = tid; i < D; i += 256) sum += xr[i];
#pragma unroll
  for (int off = 16; off; off >>= 1) sum += __shfl_xor(sum, off);
  if ((tid & 31) == 0) red[tid >> 5] = sum;
  __syncthreads();
  if (tid == 0) { float t = 0.f; for (int i = 0; i < 8; ++i) t += red[i]; smean = t / D; }
  __syncthreads();
  float mean = smean;

  float vs = 0.f;
  for (int i = tid; i < D; i += 256) { float d = xr[i] - mean; vs += d * d; }
#pragma unroll
  for (int off = 16; off; off >>= 1) vs += __shfl_xor(vs, off);
  if ((tid & 31) == 0) red[tid >> 5] = vs;
  __syncthreads();
  if (tid == 0) { float t = 0.f; for (int i = 0; i < 8; ++i) t += red[i]; svar = t / D; }
  __syncthreads();
  float inv = rsqrtf(svar + 1e-6f);

  for (int i = tid; i < D; i += 256) {
    float val = (xr[i] - mean) * inv * gam[i] + bet[i];
    if (doRelu) val = fmaxf(val, 0.f);
    yr[i] = val;
  }
}

// ---------------------------------------------------------------------------
// Launch. Inputs (setup_inputs dict order, params flattened in key order):
//  0 node_feats  1 edge_feats  2 q_w  3 q_b  4 k_w  5 k_b  6 v_w  7 v_b
//  8 score_w  9 score_b  10 e1_w 11 e1_b 12 e_ln_g 13 e_ln_b 14 e2_w 15 e2_b
//  16 fn_g 17 fn_b
// d_out = [out (B*S*D) | edge_feats (B*S*D)]
// ---------------------------------------------------------------------------
extern "C" void kernel_launch(void* const* d_in, const int* in_sizes, int n_in,
                              void* d_out, int out_size, void* d_ws, size_t ws_size,
                              hipStream_t stream) {
  const float* node    = (const float*)d_in[0];
  const float* edge    = (const float*)d_in[1];
  const float* q_w     = (const float*)d_in[2];
  const float* q_b     = (const float*)d_in[3];
  const float* k_w     = (const float*)d_in[4];
  const float* k_b     = (const float*)d_in[5];
  const float* v_w     = (const float*)d_in[6];
  const float* v_b     = (const float*)d_in[7];
  const float* score_w = (const float*)d_in[8];
  const float* score_b = (const float*)d_in[9];
  const float* e1_w    = (const float*)d_in[10];
  const float* e1_b    = (const float*)d_in[11];
  const float* e_ln_g  = (const float*)d_in[12];
  const float* e_ln_b  = (const float*)d_in[13];
  const float* e2_w    = (const float*)d_in[14];
  const float* e2_b    = (const float*)d_in[15];
  const float* fn_g    = (const float*)d_in[16];
  const float* fn_b    = (const float*)d_in[17];

  const int B = 2, S = 2048, D = 1024, H = 16, C = 32;
  const int M = B * S;

  float* ws = (float*)d_ws;
  size_t off = 0;
  float* qbuf = ws + off; off += (size_t)M * D;
  float* kbuf = ws + off; off += (size_t)M * D;
  float* vbuf = ws + off; off += (size_t)M * D;
  float* attn = ws + off; off += (size_t)M * D;
  float* km   = ws + off; off += (size_t)B * C * H * 64;
  float* scores = ws + off; off += (size_t)M;
  unsigned* selm = (unsigned*)(ws + off); off += (size_t)B * H * S;
  int* cnt  = (int*)(ws + off); off += 8;
  int* keff = cnt + 4;
  float* h1 = qbuf;  // reuse after attention
  float* h3 = kbuf;  // reuse after attention

  // dynamic top-k (device-side, graph-capture safe)
  score_kernel<<<(M + 7) / 8, 256, 0, stream>>>(node, score_w, score_b, scores, D, M);
  stats_kernel<<<B, 256, 0, stream>>>(scores, S, cnt);
  finalize_topk_kernel<<<1, 1, 0, stream>>>(cnt, B, C, keff);

  // q/k/v projections (WMMA)
  dim3 gg(M / 128, D / 64);
  gemm_f16wmma<<<gg, 128, 0, stream>>>(node, q_w, q_b, qbuf, M, D, D);
  gemm_f16wmma<<<gg, 128, 0, stream>>>(node, k_w, k_b, kbuf, M, D, D);
  gemm_f16wmma<<<gg, 128, 0, stream>>>(node, v_w, v_b, vbuf, M, D, D);

  // routing
  chunkmean_kernel<<<(B * C * H * 64 + 255) / 256, 256, 0, stream>>>(kbuf, km, B, S, H);
  gate_topk_kernel<<<(B * H * S + 3) / 4, 128, 0, stream>>>(qbuf, km, keff, selm, B, S, H);

  // block-sparse flash attention (WMMA)
  moba_flash_kernel<<<(B * H * (S / 16)) / 4, 128, 0, stream>>>(qbuf, kbuf, vbuf, selm,
                                                                attn, B, S, H);

  // task head: e1 -> relu(LN) -> e2 -> LN
  gemm_f16wmma<<<gg, 128, 0, stream>>>(attn, e1_w, e1_b, h1, M, D, D);
  layernorm_kernel<<<M, 256, 0, stream>>>(h1, e_ln_g, e_ln_b, h1, D, 1);
  gemm_f16wmma<<<gg, 128, 0, stream>>>(h1, e2_w, e2_b, h3, M, D, D);
  layernorm_kernel<<<M, 256, 0, stream>>>(h3, fn_g, fn_b, (float*)d_out, D, 0);

  // second tuple element: edge_feats passthrough
  hipMemcpyAsync((float*)d_out + (size_t)M * D, edge, (size_t)M * D * sizeof(float),
                 hipMemcpyDeviceToDevice, stream);
}
